// SpatialPooler_14173392077106
// MI455X (gfx1250) — 8-bit matrix operand order repaired and checked by reading, compiled, not run on hardware
//
#include <hip/hip_runtime.h>
#include <hip/hip_bf16.h>

typedef __attribute__((ext_vector_type(16))) int   v16i;
typedef __attribute__((ext_vector_type(8)))  float v8f;

#define BATCH     8192
#define IN_DIM    2048
#define OUT_DIM   4096
#define TOPK      82
#define BOOST_STR 100.0f

#define BM 128
#define BN 128
#define BK 128

__device__ __forceinline__ unsigned char f2fp8bin(float v) {
    return v != 0.0f ? (unsigned char)0x38 : (unsigned char)0x00;
}

__device__ __forceinline__ void async_copy64(unsigned ldsOff, const unsigned char* g) {
    unsigned long long ga = (unsigned long long)(uintptr_t)g;
    asm volatile(
        "global_load_async_to_lds_b128 %0, %1, off\n\t"
        "global_load_async_to_lds_b128 %0, %1, off offset:16\n\t"
        "global_load_async_to_lds_b128 %0, %1, off offset:32\n\t"
        "global_load_async_to_lds_b128 %0, %1, off offset:48"
        :: "v"(ldsOff), "v"(ga) : "memory");
}

__device__ __forceinline__ void wait_async0() {
    asm volatile("s_wait_asynccnt 0x0" ::: "memory");
}

__global__ __launch_bounds__(256) void boost_kernel(const float* __restrict__ avg,
                                                    float* __restrict__ boost) {
    __shared__ float red[256];
    int t = threadIdx.x;
    float s = 0.0f;
    for (int i = t; i < OUT_DIM; i += 256) s += avg[i];
    red[t] = s;
    __syncthreads();
    for (int off = 128; off > 0; off >>= 1) {
        if (t < off) red[t] += red[t + off];
        __syncthreads();
    }
    float total = red[0];
    for (int i = t; i < OUT_DIM; i += 256) {
        float a = avg[i];
        float neighbor = (total - a) / (float)(OUT_DIM - 1);
        boost[i] = expf(-BOOST_STR * (a - neighbor));
    }
}

__global__ __launch_bounds__(256) void pack_x_kernel(const float* __restrict__ in,
                                                     unsigned char* __restrict__ out) {
    size_t idx = ((size_t)blockIdx.x * 256 + threadIdx.x) * 16;
    const float4* p = (const float4*)(in + idx);
    unsigned int w[4];
#pragma unroll
    for (int q = 0; q < 4; ++q) {
        float4 v = p[q];
        w[q] = (unsigned int)f2fp8bin(v.x)
             | ((unsigned int)f2fp8bin(v.y) << 8)
             | ((unsigned int)f2fp8bin(v.z) << 16)
             | ((unsigned int)f2fp8bin(v.w) << 24);
    }
    *(uint4*)(out + idx) = make_uint4(w[0], w[1], w[2], w[3]);
}

__global__ __launch_bounds__(256) void pack_connT_kernel(const float* __restrict__ conn,
                                                         unsigned char* __restrict__ cq) {
    __shared__ unsigned char tile[64][68];
    int n0 = blockIdx.x * 64;
    int k0 = blockIdx.y * 64;
    int t  = threadIdx.x;

    int lr  = t >> 4;
    int lc4 = (t & 15) * 4;
#pragma unroll
    for (int i = 0; i < 4; ++i) {
        int kr = lr + i * 16;
        float4 v = *(const float4*)(conn + (size_t)(k0 + kr) * OUT_DIM + n0 + lc4);
        tile[kr][lc4 + 0] = f2fp8bin(v.x);
        tile[kr][lc4 + 1] = f2fp8bin(v.y);
        tile[kr][lc4 + 2] = f2fp8bin(v.z);
        tile[kr][lc4 + 3] = f2fp8bin(v.w);
    }
    __syncthreads();

    int n  = t >> 2;
    int kc = (t & 3) * 16;
    unsigned int w[4];
#pragma unroll
    for (int q = 0; q < 4; ++q) {
        w[q] = (unsigned int)tile[kc + q * 4 + 0][n]
             | ((unsigned int)tile[kc + q * 4 + 1][n] << 8)
             | ((unsigned int)tile[kc + q * 4 + 2][n] << 16)
             | ((unsigned int)tile[kc + q * 4 + 3][n] << 24);
    }
    unsigned char* dst = cq + (size_t)(n0 + n) * IN_DIM + k0 + kc;
    *(uint4*)dst = make_uint4(w[0], w[1], w[2], w[3]);
}

__global__ __launch_bounds__(256) void gemm_fp8_kernel(const unsigned char* __restrict__ Aq,
                                                       const unsigned char* __restrict__ Bq,
                                                       const float* __restrict__ boost,
                                                       float* __restrict__ out) {
    __shared__ __align__(16) unsigned char As[2][BM * BK];
    __shared__ __align__(16) unsigned char Bs[2][BN * BK];

    int t    = threadIdx.x;
    int wave = t >> 5;
    int lane = t & 31;
    int l16  = lane & 15;
    int hsel = lane >> 4;

    int mBase = blockIdx.y * BM;
    int nBase = blockIdx.x * BN;

    int wm = (wave & 3) * 2;
    int wn = (wave >> 2) * 4;

    v8f acc[2][4];
#pragma unroll
    for (int mi = 0; mi < 2; ++mi)
#pragma unroll
        for (int ni = 0; ni < 4; ++ni)
            acc[mi][ni] = (v8f){0.f, 0.f, 0.f, 0.f, 0.f, 0.f, 0.f, 0.f};

    int ldRow = t >> 1;
    int ldSeg = (t & 1) * 64;
    unsigned aOff[2], bOff[2];
    aOff[0] = (unsigned)(uintptr_t)&As[0][ldRow * BK + ldSeg];
    aOff[1] = (unsigned)(uintptr_t)&As[1][ldRow * BK + ldSeg];
    bOff[0] = (unsigned)(uintptr_t)&Bs[0][ldRow * BK + ldSeg];
    bOff[1] = (unsigned)(uintptr_t)&Bs[1][ldRow * BK + ldSeg];
    const unsigned char* srcA = Aq + (size_t)(mBase + ldRow) * IN_DIM + ldSeg;
    const unsigned char* srcB = Bq + (size_t)(nBase + ldRow) * IN_DIM + ldSeg;

    const int NCHUNK = IN_DIM / BK;

    async_copy64(aOff[0], srcA);
    async_copy64(bOff[0], srcB);

    for (int i = 0; i < NCHUNK; ++i) {
        int cur = i & 1;
        wait_async0();
        __syncthreads();
        if (i + 1 < NCHUNK) {
            async_copy64(aOff[cur ^ 1], srcA + (size_t)(i + 1) * BK);
            async_copy64(bOff[cur ^ 1], srcB + (size_t)(i + 1) * BK);
        }

        const unsigned char* Ab = &As[cur][0];
        const unsigned char* Bb = &Bs[cur][0];

        v16i afrag[2];
#pragma unroll
        for (int mi = 0; mi < 2; ++mi) {
            int arow = ((wm + mi) * 16 + l16) * BK;
#pragma unroll
            for (int c = 0; c < 4; ++c) {
                int4 d = *(const int4*)&Ab[arow + c * 32 + hsel * 16];
                afrag[mi][4 * c + 0] = d.x;
                afrag[mi][4 * c + 1] = d.y;
                afrag[mi][4 * c + 2] = d.z;
                afrag[mi][4 * c + 3] = d.w;
            }
        }
        v16i bfrag[4];
#pragma unroll
        for (int ni = 0; ni < 4; ++ni) {
            int brow = ((wn + ni) * 16 + l16) * BK;
#pragma unroll
            for (int g = 0; g < 4; ++g) {
                int4 d = *(const int4*)&Bb[brow + g * 32 + hsel * 16];
                bfrag[ni][4 * g + 0] = d.x;
                bfrag[ni][4 * g + 1] = d.y;
                bfrag[ni][4 * g + 2] = d.z;
                bfrag[ni][4 * g + 3] = d.w;
            }
        }
#pragma unroll
        for (int mi = 0; mi < 2; ++mi)
#pragma unroll
            for (int ni = 0; ni < 4; ++ni)
                acc[mi][ni] = __builtin_amdgcn_wmma_f32_16x16x128_fp8_fp8(
                    afrag[mi], bfrag[ni], (short)0, acc[mi][ni], false, false);
    }

#pragma unroll
    for (int ni = 0; ni < 4; ++ni) {
        int col = nBase + (wn + ni) * 16 + l16;
        float bf = boost[col];
#pragma unroll
        for (int mi = 0; mi < 2; ++mi) {
#pragma unroll
            for (int r = 0; r < 8; ++r) {
                int rowg = mBase + (wm + mi) * 16 + r + hsel * 8;
                out[(size_t)rowg * OUT_DIM + col] = acc[mi][ni][r] * bf;
            }
        }
    }
}

__global__ __launch_bounds__(256) void topk_kernel(float* __restrict__ out) {
    __shared__ unsigned int keys[OUT_DIM];
    __shared__ int hist[256];
    __shared__ int scanbuf[256];
    __shared__ unsigned int s_prefix;
    __shared__ int s_kprime;

    int t = threadIdx.x;
    float* rowp = out + (size_t)blockIdx.x * OUT_DIM;

    for (int i = t; i < OUT_DIM; i += 256) keys[i] = __float_as_uint(rowp[i]);
    if (t == 0) { s_prefix = 0u; s_kprime = TOPK; }
    __syncthreads();

    for (int pass = 0; pass < 4; ++pass) {
        int shift = 24 - pass * 8;
        unsigned int pmask = (pass == 0) ? 0u : (0xFFFFFFFFu << (shift + 8));
        hist[t] = 0;
        __syncthreads();
        unsigned int prefix = s_prefix;
        for (int i = t; i < OUT_DIM; i += 256) {
            unsigned int k = keys[i];
            if ((k & pmask) == prefix)
                atomicAdd(&hist[(k >> shift) & 0xFF], 1);
        }
        __syncthreads();
        if (t == 0) {
            int kp = s_kprime;
            int b = 255;
            for (; b > 0; --b) {
                int c = hist[b];
                if (kp <= c) break;
                kp -= c;
            }
            s_prefix = prefix | ((unsigned int)b << shift);
            s_kprime = kp;
        }
        __syncthreads();
    }

    unsigned int T = s_prefix;
    int remaining  = s_kprime;

    int base = t * 16;
    int localEq = 0;
#pragma unroll
    for (int j = 0; j < 16; ++j) localEq += (keys[base + j] == T) ? 1 : 0;
    scanbuf[t] = localEq;
    __syncthreads();
    for (int off = 1; off < 256; off <<= 1) {
        int self = scanbuf[t];
        int add  = (t >= off) ? scanbuf[t - off] : 0;
        __syncthreads();
        scanbuf[t] = self + add;
        __syncthreads();
    }
    int rank = scanbuf[t] - localEq;

#pragma unroll
    for (int j = 0; j < 16; ++j) {
        unsigned int k = keys[base + j];
        float o;
        if (k > T) {
            o = 1.0f;
        } else if (k == T) {
            o = (rank < remaining) ? 1.0f : 0.0f;
            ++rank;
        } else {
            o = 0.0f;
        }
        rowp[base + j] = o;
    }
}

extern "C" void kernel_launch(void* const* d_in, const int* in_sizes, int n_in,
                              void* d_out, int out_size, void* d_ws, size_t ws_size,
                              hipStream_t stream) {
    const float* x    = (const float*)d_in[0];
    const float* conn = (const float*)d_in[1];
    const float* avg  = (const float*)d_in[2];
    float* out        = (float*)d_out;

    unsigned char* ws = (unsigned char*)d_ws;
    unsigned char* xq = ws;
    unsigned char* cq = ws + (size_t)BATCH * IN_DIM;
    float* boost = (float*)(ws + (size_t)BATCH * IN_DIM + (size_t)OUT_DIM * IN_DIM);

    boost_kernel<<<1, 256, 0, stream>>>(avg, boost);

    pack_x_kernel<<<(int)(((size_t)BATCH * IN_DIM) / (256 * 16)), 256, 0, stream>>>(x, xq);

    pack_connT_kernel<<<dim3(OUT_DIM / 64, IN_DIM / 64), 256, 0, stream>>>(conn, cq);

    gemm_fp8_kernel<<<dim3(OUT_DIM / BN, BATCH / BM), 256, 0, stream>>>(xq, cq, boost, out);

    topk_kernel<<<BATCH, 256, 0, stream>>>(out);
}
